// ImprovedAttention_996432412850
// MI455X (gfx1250) — compile-verified, matrix operand order checked by reading, not run on hardware
//
#include <hip/hip_runtime.h>

// ---------------------------------------------------------------------------
// MI455X (gfx1250) fused attention: bf16 WMMA everywhere, f32 accumulate.
// ---------------------------------------------------------------------------

typedef __attribute__((ext_vector_type(16))) __bf16        v16bf;
typedef __attribute__((ext_vector_type(8)))  float         v8f;
typedef __attribute__((ext_vector_type(4)))  unsigned int  v4u;

#define B_    2
#define T_    2048
#define C_    1024
#define H_    16
#define D_    64

union BfVec { v16bf bf; v4u u[2]; };

// Load a 16x32 bf16 WMMA fragment (A layout; also B when the operand is
// stored K-major, i.e. Bt[n][k]).  Per ISA 7.12.2: lane = row (&15),
// lane>>4 selects the K half; VGPRs 0-3 hold K = half*8..+7, VGPRs 4-7 hold
// K = 16+half*8..+7  ->  exactly two global_load_b128 per lane.
__device__ inline v16bf load_mk(const __bf16* base, int ld, int lane) {
  int row  = lane & 15;
  int half = (lane >> 4) & 1;
  const __bf16* p = base + (size_t)row * ld + half * 8;
  BfVec t;
  t.u[0] = *reinterpret_cast<const v4u*>(p);
  t.u[1] = *reinterpret_cast<const v4u*>(p + 16);
  return t.bf;
}

__device__ inline v8f wmma_bf16(v16bf a, v16bf b, v8f c) {
  return __builtin_amdgcn_wmma_f32_16x16x32_bf16(false, a, false, b,
                                                 (short)0, c, false, false);
}

// ---------------------------------------------------------------------------
// 1) fp32 -> bf16 elementwise convert
// ---------------------------------------------------------------------------
__global__ void cvt_kernel(const float* __restrict__ in,
                           __bf16* __restrict__ out, int n) {
  int i = blockIdx.x * blockDim.x + threadIdx.x;
  if (i < n) out[i] = (__bf16)in[i];
}

// ---------------------------------------------------------------------------
// 2) W[k][n] -> Wt[n][k] bf16, 32x32 LDS tiles
// ---------------------------------------------------------------------------
__global__ void transpose_cvt_kernel(const float* __restrict__ in,
                                     __bf16* __restrict__ out, int n) {
  __shared__ float tile[32][33];
  int x  = blockIdx.x * 32 + threadIdx.x;
  int y0 = blockIdx.y * 32;
  for (int i = threadIdx.y; i < 32; i += 8)
    tile[i][threadIdx.x] = in[(size_t)(y0 + i) * n + x];
  __syncthreads();
  int xo  = y0 + threadIdx.x;
  int yo0 = blockIdx.x * 32;
  for (int i = threadIdx.y; i < 32; i += 8)
    out[(size_t)(yo0 + i) * n + xo] = (__bf16)tile[threadIdx.x][i];
}

// ---------------------------------------------------------------------------
// Wave-level GEMM helper: one wave computes a 16(M) x 64(N) tile.
// A: row-major [16 x K] (lda), Bt: K-major [64 x K] (ldb).
// ---------------------------------------------------------------------------
__device__ inline void gemm_16x64(const __bf16* Arow, int lda,
                                  const __bf16* Bt, int ldb,
                                  int K, int lane, v8f acc[4]) {
  for (int k0 = 0; k0 < K; k0 += 32) {
    v16bf a = load_mk(Arow + k0, lda, lane);
#pragma unroll
    for (int j = 0; j < 4; ++j) {
      v16bf bj = load_mk(Bt + (size_t)(j * 16) * ldb + k0, ldb, lane);
      acc[j] = wmma_bf16(a, bj, acc[j]);
    }
  }
}

// ---------------------------------------------------------------------------
// 3) QKV projection + in-register RoPE.
//    grid = (H_, BT/128, 3): z=0 -> Q, z=1 -> K (both RoPE'd, [B,H,T,D]),
//    z=2 -> V (stored transposed [B,H,D,T]).
// ---------------------------------------------------------------------------
__global__ __launch_bounds__(256)
void qkv_kernel(const __bf16* __restrict__ xb,
                const __bf16* __restrict__ wqt,
                const __bf16* __restrict__ wkt,
                const __bf16* __restrict__ wvt,
                __bf16* __restrict__ qb,
                __bf16* __restrict__ kb,
                __bf16* __restrict__ vt) {
  int lane = threadIdx.x & 31;
  int wv32 = threadIdx.x >> 5;
  int z    = blockIdx.z;
  const __bf16* wt = (z == 0) ? wqt : (z == 1) ? wkt : wvt;

  int h  = blockIdx.x;                       // head == 64-wide column block
  int m0 = blockIdx.y * 128 + wv32 * 16;     // row in [0, B*T)
  int n0 = h * 64;

  v8f acc[4] = {v8f{0.f}, v8f{0.f}, v8f{0.f}, v8f{0.f}};
  gemm_16x64(xb + (size_t)m0 * C_, C_, wt + (size_t)n0 * C_, C_, C_, lane, acc);

  int nloc = lane & 15;
  int hrow = lane >> 4;
  int bb   = m0 >> 11;                       // row block fully inside one batch

  if (z < 2) {
    __bf16* dst = ((z == 0) ? qb : kb) + ((size_t)(bb * H_ + h) * T_) * D_;
#pragma unroll
    for (int j = 0; j < 4; ++j) {
      int d = j * 16 + nloc;
      float invf = __powf(10000.0f, -(float)(d & 31) * (1.0f / 32.0f));
#pragma unroll
      for (int r = 0; r < 8; ++r) {
        int t = (m0 + hrow * 8 + r) & (T_ - 1);
        float sv, cv;
        __sincosf((float)t * invf, &sv, &cv);
        float v   = acc[j][r];
        float par = acc[j ^ 2][r];            // rotate_half partner (d +/- 32)
        float rot = (d < 32) ? -par : par;
        dst[(size_t)t * D_ + d] = (__bf16)(v * cv + rot * sv);
      }
    }
  } else {
    __bf16* dst = vt + ((size_t)(bb * H_ + h) * D_) * T_;
#pragma unroll
    for (int j = 0; j < 4; ++j)
#pragma unroll
      for (int r = 0; r < 8; ++r) {
        int t = (m0 + hrow * 8 + r) & (T_ - 1);
        dst[(size_t)(j * 16 + nloc) * T_ + t] = (__bf16)acc[j][r];
      }
  }
}

// ---------------------------------------------------------------------------
// 4) Flash attention: 1 wave = 16 query rows, key chunks of 32.
//    grid = (T/128, H, B), block = 256 (8 waves), scale = 1.0 (RoPE branch).
// ---------------------------------------------------------------------------
__global__ __launch_bounds__(256)
void attn_kernel(const __bf16* __restrict__ qb,
                 const __bf16* __restrict__ kb,
                 const __bf16* __restrict__ vt,
                 const int* __restrict__ amask,
                 const int* __restrict__ icp,
                 __bf16* __restrict__ ob) {
  __shared__ __attribute__((aligned(16))) __bf16 pbuf[8 * 16 * 32];

  int lane = threadIdx.x & 31;
  int wv32 = threadIdx.x >> 5;
  int h = blockIdx.y, b = blockIdx.z;
  int q0 = blockIdx.x * 128 + wv32 * 16;
  int causal = icp[0];

  const __bf16* qh = qb + ((size_t)(b * H_ + h) * T_) * D_;
  const __bf16* kh = kb + ((size_t)(b * H_ + h) * T_) * D_;
  const __bf16* vh = vt + ((size_t)(b * H_ + h) * D_) * T_;
  __bf16* pb = pbuf + wv32 * 512;

  int nloc = lane & 15;
  int hrow = lane >> 4;

  v16bf aq0 = load_mk(qh + (size_t)q0 * D_ + 0,  D_, lane);
  v16bf aq1 = load_mk(qh + (size_t)q0 * D_ + 32, D_, lane);

  v8f o[4] = {v8f{0.f}, v8f{0.f}, v8f{0.f}, v8f{0.f}};
  float rmax[8], rsum[8];
#pragma unroll
  for (int r = 0; r < 8; ++r) { rmax[r] = -3.0e38f; rsum[r] = 0.0f; }

  int kend = causal ? (q0 + 16) : T_;
  for (int kt0 = 0; kt0 < kend; kt0 += 32) {
    // ---- S = Q K^T for a 16q x 32k chunk (two 16x16 tiles) ----
    v8f s0 = v8f{0.f}, s1 = v8f{0.f};
    v16bf bk;
    bk = load_mk(kh + (size_t)kt0 * D_ + 0,         D_, lane); s0 = wmma_bf16(aq0, bk, s0);
    bk = load_mk(kh + (size_t)kt0 * D_ + 32,        D_, lane); s0 = wmma_bf16(aq1, bk, s0);
    bk = load_mk(kh + (size_t)(kt0 + 16) * D_ + 0,  D_, lane); s1 = wmma_bf16(aq0, bk, s1);
    bk = load_mk(kh + (size_t)(kt0 + 16) * D_ + 32, D_, lane); s1 = wmma_bf16(aq1, bk, s1);

    int tk0 = kt0 + nloc, tk1 = kt0 + 16 + nloc;
    int pad0 = amask[b * T_ + tk0];
    int pad1 = amask[b * T_ + tk1];

    // ---- online softmax (rows live on 16 lanes; shfl_xor width 16) ----
#pragma unroll
    for (int r = 0; r < 8; ++r) {
      int tq = q0 + hrow * 8 + r;
      float v0 = s0[r], v1 = s1[r];
      if (!pad0 || (causal && tk0 > tq)) v0 = -3.0e38f;
      if (!pad1 || (causal && tk1 > tq)) v1 = -3.0e38f;
      float m = fmaxf(v0, v1);
      m = fmaxf(m, __shfl_xor(m, 1, 16));
      m = fmaxf(m, __shfl_xor(m, 2, 16));
      m = fmaxf(m, __shfl_xor(m, 4, 16));
      m = fmaxf(m, __shfl_xor(m, 8, 16));
      float nm = fmaxf(rmax[r], m);
      float sc = __expf(rmax[r] - nm);
      rmax[r] = nm;
      float p0 = (v0 > -1.0e37f) ? __expf(v0 - nm) : 0.0f;
      float p1 = (v1 > -1.0e37f) ? __expf(v1 - nm) : 0.0f;
      float ps = p0 + p1;
      ps += __shfl_xor(ps, 1, 16);
      ps += __shfl_xor(ps, 2, 16);
      ps += __shfl_xor(ps, 4, 16);
      ps += __shfl_xor(ps, 8, 16);
      rsum[r] = rsum[r] * sc + ps;
      o[0][r] *= sc; o[1][r] *= sc; o[2][r] *= sc; o[3][r] *= sc;
      // stage P (bf16) in wave-private LDS, plain 16x32 row-major
      pb[(hrow * 8 + r) * 32 + nloc]      = (__bf16)p0;
      pb[(hrow * 8 + r) * 32 + 16 + nloc] = (__bf16)p1;
    }
    asm volatile("s_wait_dscnt 0" ::: "memory");   // wave-private: no barrier

    // ---- O += P @ V ----
    v16bf pa = load_mk(pb, 32, lane);
#pragma unroll
    for (int j = 0; j < 4; ++j) {
      v16bf bv = load_mk(vh + (size_t)(j * 16) * T_ + kt0, T_, lane);
      o[j] = wmma_bf16(pa, bv, o[j]);
    }
  }

  // ---- normalize, store [B, T, H*D] bf16 for the out-projection ----
#pragma unroll
  for (int r = 0; r < 8; ++r) {
    float inv = (rsum[r] > 0.0f) ? (1.0f / rsum[r]) : 0.0f;
    int tq = q0 + hrow * 8 + r;
    __bf16* dst = ob + ((size_t)(b * T_ + tq)) * C_ + h * D_;
#pragma unroll
    for (int j = 0; j < 4; ++j)
      dst[j * 16 + nloc] = (__bf16)(o[j][r] * inv);
  }
}

// ---------------------------------------------------------------------------
// 5) out = attn_out @ Wo + bo  (fp32 result)
// ---------------------------------------------------------------------------
__global__ __launch_bounds__(256)
void proj_kernel(const __bf16* __restrict__ ab,
                 const __bf16* __restrict__ wot,
                 const float* __restrict__ bias,
                 float* __restrict__ out) {
  int lane = threadIdx.x & 31;
  int wv32 = threadIdx.x >> 5;
  int m0 = blockIdx.y * 128 + wv32 * 16;
  int n0 = blockIdx.x * 64;

  v8f acc[4] = {v8f{0.f}, v8f{0.f}, v8f{0.f}, v8f{0.f}};
  gemm_16x64(ab + (size_t)m0 * C_, C_, wot + (size_t)n0 * C_, C_, C_, lane, acc);

  int nloc = lane & 15;
  int hrow = lane >> 4;
#pragma unroll
  for (int j = 0; j < 4; ++j) {
    int n = n0 + j * 16 + nloc;
    float bv = bias[n];
#pragma unroll
    for (int r = 0; r < 8; ++r)
      out[(size_t)(m0 + hrow * 8 + r) * C_ + n] = acc[j][r] + bv;
  }
}

// ---------------------------------------------------------------------------
extern "C" void kernel_launch(void* const* d_in, const int* in_sizes, int n_in,
                              void* d_out, int out_size, void* d_ws, size_t ws_size,
                              hipStream_t stream) {
  const float* x  = (const float*)d_in[0];
  const int*   am = (const int*)  d_in[1];
  const float* wq = (const float*)d_in[2];
  const float* wk = (const float*)d_in[3];
  const float* wv = (const float*)d_in[4];
  const float* wo = (const float*)d_in[5];
  const float* bo = (const float*)d_in[6];
  const int*   ic = (const int*)  d_in[7];

  char* ws = (char*)d_ws;
  const size_t MB = 1ull << 20;
  __bf16* xb  = (__bf16*)(ws + 0 * MB);   // 8 MB  : x bf16       [B*T, C]
  __bf16* wqt = (__bf16*)(ws + 8 * MB);   // 2 MB  : Wq^T bf16    [C, C]
  __bf16* wkt = (__bf16*)(ws + 10 * MB);  // 2 MB
  __bf16* wvt = (__bf16*)(ws + 12 * MB);  // 2 MB
  __bf16* wot = (__bf16*)(ws + 14 * MB);  // 2 MB
  __bf16* qb  = (__bf16*)(ws + 16 * MB);  // 8 MB  : Q rope bf16  [B,H,T,D]
  __bf16* kb  = (__bf16*)(ws + 24 * MB);  // 8 MB  : K rope bf16  [B,H,T,D]
  __bf16* vtb = (__bf16*)(ws + 32 * MB);  // 8 MB  : V^T bf16     [B,H,D,T]
  __bf16* ob  = (__bf16*)(ws + 40 * MB);  // 8 MB  : attn out     [B,T,C]

  const int NX = B_ * T_ * C_;  // 4,194,304
  cvt_kernel<<<(NX + 255) / 256, 256, 0, stream>>>(x, xb, NX);

  dim3 tb(32, 8), tg(C_ / 32, C_ / 32);
  transpose_cvt_kernel<<<tg, tb, 0, stream>>>(wq, wqt, C_);
  transpose_cvt_kernel<<<tg, tb, 0, stream>>>(wk, wkt, C_);
  transpose_cvt_kernel<<<tg, tb, 0, stream>>>(wv, wvt, C_);
  transpose_cvt_kernel<<<tg, tb, 0, stream>>>(wo, wot, C_);

  qkv_kernel<<<dim3(H_, (B_ * T_) / 128, 3), 256, 0, stream>>>(
      xb, wqt, wkt, wvt, qb, kb, vtb);

  attn_kernel<<<dim3(T_ / 128, H_, B_), 256, 0, stream>>>(
      qb, kb, vtb, am, ic, ob);

  proj_kernel<<<dim3(C_ / 64, (B_ * T_) / 128), 256, 0, stream>>>(
      ob, wot, bo, (float*)d_out);

  (void)in_sizes; (void)n_in; (void)out_size; (void)ws_size;
}
